// AttLearner_29703993819980
// MI455X (gfx1250) — matrix operand order repaired and checked by reading, not compiled, not run on hardware
//
#include <hip/hip_runtime.h>

typedef __attribute__((ext_vector_type(16))) _Float16 v16h;
typedef __attribute__((ext_vector_type(8)))  _Float16 v8h;
typedef __attribute__((ext_vector_type(8)))  float    v8f;

#define KDIM 512
#define TILE 128
#define LDA  40   // 32 f16 + 8 f16 pad -> 80-byte row stride, 16B aligned

// CDNA5 async copy: global memory -> LDS, per-lane, tracked by ASYNCcnt.
// lds_off = LDS byte address (low 32 bits of shared-aperture pointer).
__device__ __forceinline__ void async_copy_b128(unsigned lds_off, const void* gaddr) {
    asm volatile("global_load_async_to_lds_b128 %0, %1, off"
                 :: "v"(lds_off), "v"(gaddr) : "memory");
}

// ---------------------------------------------------------------------------
// Pass 1: x = relu(features * w0) * w1 ; z = x / max(||x||2, eps) ; store f16
// ---------------------------------------------------------------------------
__global__ __launch_bounds__(256) void prep_normalize(
    const float* __restrict__ F, const float* __restrict__ W,
    _Float16* __restrict__ Z)
{
    __shared__ float red[256];
    const int row = blockIdx.x;
    const int t   = threadIdx.x;

    const float* frow = F + (size_t)row * KDIM;

    float x0 = frow[t] * W[t];
    x0 = fmaxf(x0, 0.0f) * W[KDIM + t];
    float x1 = frow[t + 256] * W[t + 256];
    x1 = fmaxf(x1, 0.0f) * W[KDIM + t + 256];

    red[t] = x0 * x0 + x1 * x1;
    __syncthreads();
#pragma unroll
    for (int s = 128; s > 0; s >>= 1) {
        if (t < s) red[t] += red[t + s];
        __syncthreads();
    }
    const float inv = 1.0f / fmaxf(sqrtf(red[0]), 1e-12f);

    _Float16* zrow = Z + (size_t)row * KDIM;
    zrow[t]       = (_Float16)(x0 * inv);
    zrow[t + 256] = (_Float16)(x1 * inv);
}

// ---------------------------------------------------------------------------
// Pass 2: out = relu(Z * Z^T). 128x128 tile / block, 8 waves, double-buffered
// LDS staged with GLOBAL_LOAD_ASYNC_TO_LDS_B128 (ASYNCcnt pipeline).
// ---------------------------------------------------------------------------
__global__ __launch_bounds__(256) void gemm_zzt_relu(
    const _Float16* __restrict__ Z, float* __restrict__ out, int N)
{
    __shared__ _Float16 As[2][TILE * LDA];
    __shared__ _Float16 Bs[2][TILE * LDA];

    const int t    = threadIdx.x;
    const int lane = t & 31;
    const int wave = t >> 5;
    const int wm   = wave & 3;   // rows wm*32 .. wm*32+31
    const int wn   = wave >> 2;  // cols wn*64 .. wn*64+63
    const int lm   = lane & 15;
    const int kh   = lane >> 4;  // K-half selector (ISA 7.12.2)

    const int bm = blockIdx.x * TILE;
    const int bn = blockIdx.y * TILE;

    // Staging assignment: 2 x 16B chunks per tile per thread.
    const int row0 = t >> 2;          // 0..63
    const int row1 = row0 + 64;       // 64..127
    const int seg  = (t & 3) * 8;     // f16 offset within 32-wide row slab

    const _Float16* gA0 = Z + (size_t)(bm + row0) * KDIM + seg;
    const _Float16* gA1 = Z + (size_t)(bm + row1) * KDIM + seg;
    const _Float16* gB0 = Z + (size_t)(bn + row0) * KDIM + seg;
    const _Float16* gB1 = Z + (size_t)(bn + row1) * KDIM + seg;

    unsigned ldsA0[2], ldsA1[2], ldsB0[2], ldsB1[2];
#pragma unroll
    for (int b = 0; b < 2; ++b) {
        ldsA0[b] = (unsigned)(uintptr_t)&As[b][row0 * LDA + seg];
        ldsA1[b] = (unsigned)(uintptr_t)&As[b][row1 * LDA + seg];
        ldsB0[b] = (unsigned)(uintptr_t)&Bs[b][row0 * LDA + seg];
        ldsB1[b] = (unsigned)(uintptr_t)&Bs[b][row1 * LDA + seg];
    }

    v8f acc[2][4];
#pragma unroll
    for (int i = 0; i < 2; ++i)
#pragma unroll
        for (int j = 0; j < 4; ++j)
            acc[i][j] = (v8f){0.f, 0.f, 0.f, 0.f, 0.f, 0.f, 0.f, 0.f};

    // Prologue: stage k=0 into buffer 0.
    async_copy_b128(ldsA0[0], gA0);
    async_copy_b128(ldsA1[0], gA1);
    async_copy_b128(ldsB0[0], gB0);
    async_copy_b128(ldsB1[0], gB1);

    int cur = 0;
#pragma unroll 2
    for (int k0 = 0; k0 < KDIM; k0 += 32) {
        // All waves finished reading the buffer we are about to overwrite.
        __syncthreads();
        if (k0 + 32 < KDIM) {
            const int nxt = cur ^ 1;
            async_copy_b128(ldsA0[nxt], gA0 + k0 + 32);
            async_copy_b128(ldsA1[nxt], gA1 + k0 + 32);
            async_copy_b128(ldsB0[nxt], gB0 + k0 + 32);
            async_copy_b128(ldsB1[nxt], gB1 + k0 + 32);
            // Leave the 4 just-issued in flight; current tile's 4 are done.
            asm volatile("s_wait_asynccnt 4" ::: "memory");
        } else {
            asm volatile("s_wait_asynccnt 0" ::: "memory");
        }
        __syncthreads();  // everyone's current-tile data visible in LDS

        const _Float16* as = As[cur];
        const _Float16* bs = Bs[cur];

        // A 16x32 frag: lane holds M=lm; K chunks {kh*8..+7} and {16+kh*8..+7}
        v16h a[2];
#pragma unroll
        for (int i = 0; i < 2; ++i) {
            const int row = wm * 32 + i * 16 + lm;
            v8h lo = *(const v8h*)(&as[row * LDA + kh * 8]);
            v8h hi = *(const v8h*)(&as[row * LDA + 16 + kh * 8]);
            a[i] = __builtin_shufflevector(lo, hi, 0, 1, 2, 3, 4, 5, 6, 7,
                                           8, 9, 10, 11, 12, 13, 14, 15);
        }
        // B 32x16 frag: lane holds N=lm; K chunks {kh*8..+7} and {16+kh*8..+7}
        v16h b[4];
#pragma unroll
        for (int j = 0; j < 4; ++j) {
            const int row = wn * 64 + j * 16 + lm;  // B col n == Z row n
            v8h lo = *(const v8h*)(&bs[row * LDA + kh * 8]);
            v8h hi = *(const v8h*)(&bs[row * LDA + 16 + kh * 8]);
            b[j] = __builtin_shufflevector(lo, hi, 0, 1, 2, 3, 4, 5, 6, 7,
                                           8, 9, 10, 11, 12, 13, 14, 15);
        }

#pragma unroll
        for (int i = 0; i < 2; ++i)
#pragma unroll
            for (int j = 0; j < 4; ++j)
                acc[i][j] = __builtin_amdgcn_wmma_f32_16x16x32_f16(
                    false, a[i], false, b[j], (short)0, acc[i][j], false, false);

        cur ^= 1;
    }

    // Store with final ReLU. C/D layout: VGPR r -> M = r + 8*kh, N = lm.
#pragma unroll
    for (int i = 0; i < 2; ++i) {
        const int mbase = bm + wm * 32 + i * 16 + 8 * kh;
#pragma unroll
        for (int j = 0; j < 4; ++j) {
            const int n = bn + wn * 64 + j * 16 + lm;
#pragma unroll
            for (int r = 0; r < 8; ++r) {
                float v = acc[i][j][r];
                out[(size_t)(mbase + r) * N + n] = v > 0.0f ? v : 0.0f;
            }
        }
    }
}

// ---------------------------------------------------------------------------
extern "C" void kernel_launch(void* const* d_in, const int* in_sizes, int n_in,
                              void* d_out, int out_size, void* d_ws, size_t ws_size,
                              hipStream_t stream) {
    const float* features = (const float*)d_in[0];   // [8192, 512] f32
    const float* w        = (const float*)d_in[1];   // [2, 512] f32
    float*       out      = (float*)d_out;           // [8192, 8192] f32
    _Float16*    Z        = (_Float16*)d_ws;         // [8192, 512] f16 (8 MB)

    const int N = in_sizes[0] / KDIM;                // 8192 rows

    prep_normalize<<<N, 256, 0, stream>>>(features, w, Z);

    dim3 grid(N / TILE, N / TILE);                   // 64 x 64 blocks
    gemm_zzt_relu<<<grid, 256, 0, stream>>>(Z, out, N);
}
